// TopKRouter_70188355551819
// MI455X (gfx1250) — matrix operand order repaired and checked by reading, not compiled, not run on hardware
//
#include <hip/hip_runtime.h>
#include <hip/hip_bf16.h>
#include <math.h>

typedef __attribute__((ext_vector_type(16))) __bf16 v16bf;
typedef __attribute__((ext_vector_type(8)))  __bf16 v8bf;
typedef __attribute__((ext_vector_type(8)))  float  v8f;
typedef __attribute__((ext_vector_type(4)))  float  v4f;

#define HDIM 4096
#define EDIM 64
#define TOKENS 16384            // B*S = 4*4096
#define TOK_PER_WAVE 16
#define WAVES_PER_BLOCK 4
#define TOK_PER_BLOCK (TOK_PER_WAVE * WAVES_PER_BLOCK)   // 64
#define NBLOCKS (TOKENS / TOK_PER_BLOCK)                 // 256
#define LDS_PITCH 68            // logits row pad

#define KSTAGE 128                          // K staged per double-buffer step
#define NSTAGES (HDIM / KSTAGE)             // 32
#define BROW_BYTES (KSTAGE * 2)             // 256B of bf16 per expert row
#define BROW_PITCH 272                      // 256 + 16 pad -> 2-way-max bank use
#define BBUF_BYTES (EDIM * BROW_PITCH)      // 17408

// ---------------- gate_w fp32 -> bf16 pre-conversion ----------------
__global__ void cvt_gate_kernel(const float* __restrict__ g,
                                __bf16* __restrict__ gbf, int n) {
    int i = blockIdx.x * blockDim.x + threadIdx.x;
    if (i < n) gbf[i] = (__bf16)g[i];
}

// ---------------- main router kernel ----------------
__global__ __launch_bounds__(128) void router_wmma_kernel(
    const float* __restrict__ x,          // [16384, 4096] fp32
    const __bf16* __restrict__ gbf,       // [64, 4096] bf16
    int*   __restrict__ out_idx,          // [16384, 2]
    float* __restrict__ out_scr,          // [16384, 2]
    float* __restrict__ partials)         // [NBLOCKS] z-loss partials
{
    __shared__ __align__(16) unsigned char bstage[2][BBUF_BYTES];
    __shared__ float lds_log[WAVES_PER_BLOCK][TOK_PER_WAVE][LDS_PITCH];
    __shared__ float red[WAVES_PER_BLOCK][TOK_PER_WAVE];

    const int wave = threadIdx.x >> 5;
    const int lane = threadIdx.x & 31;
    const int m    = lane & 15;
    const int hi   = lane >> 4;

    const int tok_base = blockIdx.x * TOK_PER_BLOCK + wave * TOK_PER_WAVE;

    // A mapping (ISA 16-bit A 16x32): lane holds token row m,
    // K runs [8*hi,+8) and [16+8*hi,+8) inside each 32-chunk.
    const float* xrow = x + (size_t)(tok_base + m) * HDIM + 8 * hi;

    // --- async stage: wave w copies expert rows [16w,16w+16), 256B each,
    //     as 8 wave-wide b128 async copies (512B per instruction). ---
    const char* gbytes = (const char*)gbf;
    auto stage = [&](int s, int buf) {
        int k0 = s * KSTAGE;
        #pragma unroll
        for (int i = 0; i < 8; ++i) {
            int f   = i * 512 + lane * 16;        // 0..4095
            int row = f >> 8;                     // 0..15 (within wave's slice)
            int col = f & 255;
            const char* src = gbytes +
                ((size_t)(wave * 16 + row) * HDIM + (size_t)k0) * 2 + col;
            unsigned dst = (unsigned)(size_t)
                (&bstage[buf][(wave * 16 + row) * BROW_PITCH + col]);
            asm volatile("global_load_async_to_lds_b128 %0, %1, off"
                         :: "v"(dst), "v"(src) : "memory");
        }
    };

    v8f c0 = {}, c1 = {}, c2 = {}, c3 = {};

    stage(0, 0);

    #pragma unroll 1
    for (int s = 0; s < NSTAGES; ++s) {
        const int cur = s & 1;
        if (s + 1 < NSTAGES) {
            stage(s + 1, 1 - cur);
            asm volatile("s_wait_asynccnt 0x8" ::: "memory"); // stage s landed
        } else {
            asm volatile("s_wait_asynccnt 0x0" ::: "memory");
        }
        __syncthreads();   // all waves' slices of stage s visible

        const unsigned char* bbuf = &bstage[cur][0];

        #pragma unroll
        for (int kk = 0; kk < 4; ++kk) {
            const int k0 = s * KSTAGE + kk * 32;

            // ---- A: fp32 from HBM, 2x 32B contiguous runs ----
            v4f a0 = *(const v4f*)(xrow + k0);
            v4f a1 = *(const v4f*)(xrow + k0 + 4);
            v4f a2 = *(const v4f*)(xrow + k0 + 16);
            v4f a3 = *(const v4f*)(xrow + k0 + 20);

            v16bf a;
            a[0]  = (__bf16)a0.x;  a[1]  = (__bf16)a0.y;
            a[2]  = (__bf16)a0.z;  a[3]  = (__bf16)a0.w;
            a[4]  = (__bf16)a1.x;  a[5]  = (__bf16)a1.y;
            a[6]  = (__bf16)a1.z;  a[7]  = (__bf16)a1.w;
            a[8]  = (__bf16)a2.x;  a[9]  = (__bf16)a2.y;
            a[10] = (__bf16)a2.z;  a[11] = (__bf16)a2.w;
            a[12] = (__bf16)a3.x;  a[13] = (__bf16)a3.y;
            a[14] = (__bf16)a3.z;  a[15] = (__bf16)a3.w;

            // ---- B: bf16 fragments from LDS (lane = expert nb*16+m,
            //      K runs [8*hi,+8) and [16+8*hi,+8) inside sub-chunk kk) ----
            #define LOAD_B(nb, dst)                                          \
                {                                                            \
                    const unsigned char* bp = bbuf +                         \
                        ((nb)*16 + m) * BROW_PITCH + kk * 64 + hi * 16;      \
                    v8bf lo = *(const v8bf*)(bp);                            \
                    v8bf hh = *(const v8bf*)(bp + 32);                       \
                    dst = __builtin_shufflevector(lo, hh, 0,1,2,3,4,5,6,7,   \
                                                  8,9,10,11,12,13,14,15);    \
                }
            v16bf b0, b1, b2, b3;
            LOAD_B(0, b0) LOAD_B(1, b1) LOAD_B(2, b2) LOAD_B(3, b3)
            #undef LOAD_B

            c0 = __builtin_amdgcn_wmma_f32_16x16x32_bf16(false, a, false, b0,
                                                         (short)0, c0, false, false);
            c1 = __builtin_amdgcn_wmma_f32_16x16x32_bf16(false, a, false, b1,
                                                         (short)0, c1, false, false);
            c2 = __builtin_amdgcn_wmma_f32_16x16x32_bf16(false, a, false, b2,
                                                         (short)0, c2, false, false);
            c3 = __builtin_amdgcn_wmma_f32_16x16x32_bf16(false, a, false, b3,
                                                         (short)0, c3, false, false);
        }
        __syncthreads();   // all waves done reading buf before it is re-staged
    }

    // ---- dump C fragments: D layout -> lane n holds M=r+8*hi at VGPR r ----
    #pragma unroll
    for (int r = 0; r < 8; ++r) {
        int mm = r + 8 * hi;
        lds_log[wave][mm][ 0 + m] = c0[r];
        lds_log[wave][mm][16 + m] = c1[r];
        lds_log[wave][mm][32 + m] = c2[r];
        lds_log[wave][mm][48 + m] = c3[r];
    }
    __syncthreads();

    // ---- per-token softmax / top-2 / ssq (lanes 0..15 of each wave) ----
    if (lane < 16) {
        const float* lrow = &lds_log[wave][lane][0];
        float m1 = -3.402823466e38f, m2 = -3.402823466e38f;
        int   i1 = 0, i2 = 0;
        float ssq = 0.0f;
        #pragma unroll 8
        for (int i = 0; i < EDIM; ++i) {
            float v = lrow[i];
            ssq += v * v;
            if (v > m1)      { m2 = m1; i2 = i1; m1 = v; i1 = i; }
            else if (v > m2) { m2 = v;  i2 = i; }
        }
        float zs = 0.0f;
        #pragma unroll 8
        for (int i = 0; i < EDIM; ++i) zs += __expf(lrow[i] - m1);
        float inv_z = 1.0f / zs;
        int gt = tok_base + lane;
        out_idx[2 * gt]     = i1;
        out_idx[2 * gt + 1] = i2;
        out_scr[2 * gt]     = inv_z;                   // exp(m1-m1)/Z
        out_scr[2 * gt + 1] = __expf(m2 - m1) * inv_z;
        red[wave][lane] = ssq;
    }
    __syncthreads();

    if (threadIdx.x == 0) {
        float s = 0.0f;
        #pragma unroll
        for (int w = 0; w < WAVES_PER_BLOCK; ++w)
            for (int t = 0; t < TOK_PER_WAVE; ++t) s += red[w][t];
        partials[blockIdx.x] = s;
    }
}

// ---------------- deterministic scalar finalize ----------------
__global__ void finalize_kernel(const float* __restrict__ partials,
                                float* __restrict__ out_scalars) {
    if (threadIdx.x == 0) {
        float s = 0.0f;
        for (int i = 0; i < NBLOCKS; ++i) s += partials[i];
        out_scalars[0] = 0.0f;                                   // aux_loss
        out_scalars[1] = s * (1.0f / ((float)TOKENS * EDIM));    // z_loss
    }
}

extern "C" void kernel_launch(void* const* d_in, const int* in_sizes, int n_in,
                              void* d_out, int out_size, void* d_ws, size_t ws_size,
                              hipStream_t stream) {
    const float* x    = (const float*)d_in[0];   // [4,4096,4096]
    const float* gate = (const float*)d_in[1];   // [64,4096]

    char*   ws       = (char*)d_ws;
    __bf16* gbf      = (__bf16*)ws;                                  // 512 KB
    float*  partials = (float*)(ws + (size_t)EDIM * HDIM * sizeof(__bf16));

    int*   out_idx     = (int*)d_out;                 // 32768 ints
    float* out_scr     = (float*)d_out + TOKENS * 2;  // 32768 floats
    float* out_scalars = (float*)d_out + TOKENS * 4;  // aux, z

    const int ngate = EDIM * HDIM;
    hipLaunchKernelGGL(cvt_gate_kernel, dim3((ngate + 255) / 256), dim3(256), 0,
                       stream, gate, gbf, ngate);
    hipLaunchKernelGGL(router_wmma_kernel, dim3(NBLOCKS), dim3(128), 0, stream,
                       x, gbf, out_idx, out_scr, partials);
    hipLaunchKernelGGL(finalize_kernel, dim3(1), dim3(32), 0, stream,
                       partials, out_scalars);
}
